// AttentionHead_41489384079839
// MI455X (gfx1250) — matrix operand order repaired and checked by reading, not compiled, not run on hardware
//
#include <hip/hip_runtime.h>

// ---------------------------------------------------------------------------
// AttentionHead (no-softmax variant): out = scale * q @ (K^T V)
//   q = x Wq + bq ; k = x Wk + bk ; v = x Wv + bv        (bf16 WMMA, f32 accum)
//   M_b = K_b^T V_b   [64x64 per batch]                  (f32, deterministic)
//   out = scale * q @ M                                  (f32)
// MI455X / gfx1250: wave32, v_wmma_f32_16x16x32_bf16, async global->LDS
// staging (ASYNCcnt) for both the x tile (raw f32) and prepacked bf16 weights.
// ---------------------------------------------------------------------------

typedef __attribute__((ext_vector_type(16))) __bf16 v16bf;
typedef __attribute__((ext_vector_type(8)))  float  v8f;

#define N_EMBD   768
#define HEADSZ   64
#define BATCH    4
#define SEQ      4096
#define NROWS    (BATCH * SEQ)      // 16384
#define NCOL     192                // q|k|v fused output columns
#define MBLK     64                 // rows per workgroup
#define PROJ_THREADS 128            // 4 waves; each wave: 16 rows x 192 cols
#define KCHUNK   32                 // WMMA K per step (bf16)
#define TCHUNK   128                // seq rows per K^T V partial
#define NCHUNKS  (SEQ / TCHUNK)     // 32

// Issue one 16-byte async global->LDS copy (gfx1250, tracked by ASYNCcnt).
__device__ __forceinline__ void async_copy_b128(unsigned lds_byte_addr,
                                                const void* gaddr)
{
    asm volatile("global_load_async_to_lds_b128 %0, %1, off"
                 :: "v"(lds_byte_addr),
                    "v"((unsigned long long)(size_t)gaddr)
                 : "memory");
}
__device__ __forceinline__ void wait_asynccnt0()
{
    asm volatile("s_wait_asynccnt 0" ::: "memory");
}

// ---------------------------------------------------------------------------
// Kernel 0: prepack weights -> bf16, transposed [col][k] so GEMM B-tiles are
// contiguous and can be async-DMA'd straight into LDS (no in-loop conversion).
// Wt[(mat*64+n)*768 + k] = bf16(Wmat[k*64 + n]);  192*768*2B = 288 KB.
// ---------------------------------------------------------------------------
__global__ __launch_bounds__(256)
void prepack_w_kernel(const float* __restrict__ Wq, const float* __restrict__ Wk,
                      const float* __restrict__ Wv, __bf16* __restrict__ Wt)
{
    const int i = blockIdx.x * 256 + threadIdx.x;     // [col][k] flat
    if (i >= NCOL * N_EMBD) return;
    const int col = i / N_EMBD;
    const int kk  = i - col * N_EMBD;
    const int mat = col >> 6;
    const int n   = col & 63;
    const float* Wp = (mat == 0) ? Wq : ((mat == 1) ? Wk : Wv);
    Wt[i] = (__bf16)Wp[(size_t)kk * HEADSZ + n];
}

// ---------------------------------------------------------------------------
// Kernel 1: fused q/k/v projection GEMM using v_wmma_f32_16x16x32_bf16.
// A-fragment k order used here (16-bit A, 16x32, B matches): lanes 0-15 hold M=lane, K
// {0..15}; lanes 16-31 hold M=lane-16, K {16..31}. x tile is
// async-staged as raw f32 and converted while building the A fragment;
// Wt tile is async-staged already in bf16 [col][k] layout.
// ---------------------------------------------------------------------------
__global__ __launch_bounds__(PROJ_THREADS)
void proj_qkv_kernel(const float* __restrict__ x,
                     const __bf16* __restrict__ Wt,
                     const float* __restrict__ bq, const float* __restrict__ bk,
                     const float* __restrict__ bv,
                     __bf16* __restrict__ qb, __bf16* __restrict__ kb,
                     __bf16* __restrict__ vb)
{
    __shared__ __attribute__((aligned(16))) float  lds_xf[MBLK * KCHUNK]; // 8 KB f32
    __shared__ __attribute__((aligned(32))) __bf16 lds_w[NCOL * KCHUNK];  // 12 KB bf16

    const int tid    = threadIdx.x;
    const int lane   = tid & 31;
    const int wid    = tid >> 5;       // 0..3
    const int laneHi = lane >> 4;      // 0/1
    const int lm     = lane & 15;
    const int rblk   = blockIdx.x * MBLK;

    v8f acc[12];
    const v8f vzero = {0.f, 0.f, 0.f, 0.f, 0.f, 0.f, 0.f, 0.f};
#pragma unroll
    for (int nt = 0; nt < 12; ++nt) acc[nt] = vzero;

    for (int kc = 0; kc < N_EMBD; kc += KCHUNK) {
        __syncthreads();   // previous iteration's LDS reads complete

        // ---- async stage x tile: 64 rows x 32 f32 = 8 KB (512 x b128) ----
#pragma unroll
        for (int j = 0; j < 4; ++j) {
            const int i   = tid + j * PROJ_THREADS;   // 0..511
            const int row = i >> 3;
            const int seg = i & 7;                    // 16B segment (4 f32)
            const float* gp = &x[(size_t)(rblk + row) * N_EMBD + kc + seg * 4];
            const unsigned lp =
                (unsigned)(size_t)(const void*)&lds_xf[row * KCHUNK + seg * 4];
            async_copy_b128(lp, gp);
        }
        // ---- async stage Wt chunk: 192 cols x 32 bf16 = 12 KB (768 x b128) ----
#pragma unroll
        for (int j = 0; j < 6; ++j) {
            const int i   = tid + j * PROJ_THREADS;   // 0..767
            const int col = i >> 2;
            const int seg = i & 3;                    // 16B segment (8 bf16)
            const __bf16* gp = &Wt[(size_t)col * N_EMBD + kc + seg * 8];
            const unsigned lp =
                (unsigned)(size_t)(const void*)&lds_w[col * KCHUNK + seg * 8];
            async_copy_b128(lp, gp);
        }
        wait_asynccnt0();
        __syncthreads();

        // ---- A fragment: convert this wave's 16 rows f32 -> bf16 ----
        const float* axf = &lds_xf[(wid * 16 + lm) * KCHUNK];
        v16bf a;
#pragma unroll
        for (int e = 0; e < 8; ++e)
            a[e] = (__bf16)axf[laneHi * 16 + e];           // K 0-7  / 16-23
#pragma unroll
        for (int e = 0; e < 8; ++e)
            a[e + 8] = (__bf16)axf[8 + laneHi * 16 + e];   // K 8-15 / 24-31

        // ---- 12 B fragments + WMMAs, depth-2 software pipeline ----
        v16bf bcur = *(const v16bf*)&lds_w[(0 * 16 + lm) * KCHUNK + laneHi * 16];
#pragma unroll
        for (int nt = 0; nt < 12; ++nt) {
            v16bf bnxt = bcur;
            if (nt < 11)
                bnxt = *(const v16bf*)&lds_w[((nt + 1) * 16 + lm) * KCHUNK +
                                             laneHi * 16];
            acc[nt] = __builtin_amdgcn_wmma_f32_16x16x32_bf16(
                false, a, false, bcur, (short)0, acc[nt], false, false);
            bcur = bnxt;
        }
    }

    // ---- epilogue: bias add (f32) then bf16 store ----
    // C/D layout: lane 0-15 -> N=lane, rows M=r; lane 16-31 -> rows M=r+8.
#pragma unroll
    for (int nt = 0; nt < 12; ++nt) {
        const int colg = nt * 16 + lm;           // 0..191
        const int mat  = colg >> 6;              // 0:q 1:k 2:v
        const int cl   = colg & 63;
        const float* bp = (mat == 0) ? bq : ((mat == 1) ? bk : bv);
        const float bias = bp[cl];
        __bf16* outp = (mat == 0) ? qb : ((mat == 1) ? kb : vb);
#pragma unroll
        for (int r = 0; r < 8; ++r) {
            const int row = rblk + wid * 16 + r + laneHi * 8;
            outp[(size_t)row * HEADSZ + cl] = (__bf16)(acc[nt][r] + bias);
        }
    }
}

// ---------------------------------------------------------------------------
// Kernel 2: per-(batch, 128-row chunk) partial of M = K^T V  (f32, no atomics
// -> bitwise deterministic). Each thread owns 1 h1 x 16 h2 outputs.
// ---------------------------------------------------------------------------
__global__ __launch_bounds__(256)
void ktv_partial_kernel(const __bf16* __restrict__ kb,
                        const __bf16* __restrict__ vb,
                        float* __restrict__ Pbuf)
{
    const int b  = blockIdx.x / NCHUNKS;
    const int c  = blockIdx.x % NCHUNKS;
    const int t0 = c * TCHUNK;
    const int h1  = threadIdx.x >> 2;          // 0..63
    const int h2b = (threadIdx.x & 3) * 16;    // 0,16,32,48

    float acc[16];
#pragma unroll
    for (int j = 0; j < 16; ++j) acc[j] = 0.f;

    const __bf16* kbase = kb + (size_t)(b * SEQ + t0) * HEADSZ;
    const __bf16* vbase = vb + (size_t)(b * SEQ + t0) * HEADSZ;
    for (int t = 0; t < TCHUNK; ++t) {
        const float kv = (float)kbase[(size_t)t * HEADSZ + h1];
        const __bf16* vr = vbase + (size_t)t * HEADSZ + h2b;
#pragma unroll
        for (int j = 0; j < 16; ++j) acc[j] += kv * (float)vr[j];
    }
    float* P = Pbuf + ((size_t)(b * NCHUNKS + c) * HEADSZ + h1) * HEADSZ + h2b;
#pragma unroll
    for (int j = 0; j < 16; ++j) P[j] = acc[j];
}

// ---------------------------------------------------------------------------
// Kernel 3: deterministic reduction of the 32 partials per batch.
// ---------------------------------------------------------------------------
__global__ __launch_bounds__(256)
void ktv_reduce_kernel(const float* __restrict__ Pbuf, float* __restrict__ Mbuf)
{
    const int idx = blockIdx.x * 256 + threadIdx.x;   // b*4096 + i, 16384 total
    const int b = idx >> 12;
    const int i = idx & 4095;
    float s = 0.f;
#pragma unroll
    for (int c = 0; c < NCHUNKS; ++c)
        s += Pbuf[((size_t)(b * NCHUNKS + c) << 12) + i];
    Mbuf[idx] = s;
}

// ---------------------------------------------------------------------------
// Kernel 4: out = scale * q @ M ; M (64x64 f32) cached in LDS, 4 rows/block.
// ---------------------------------------------------------------------------
__global__ __launch_bounds__(256)
void out_proj_kernel(const __bf16* __restrict__ qb,
                     const float* __restrict__ Mbuf,
                     float* __restrict__ out)
{
    __shared__ float Ms[HEADSZ * HEADSZ];      // 16 KB
    const int b = blockIdx.x >> 10;            // 1024 blocks per batch (4 rows each)
    for (int i = threadIdx.x; i < HEADSZ * HEADSZ; i += 256)
        Ms[i] = Mbuf[(b << 12) + i];
    __syncthreads();

    const int r  = threadIdx.x >> 6;           // 0..3
    const int h2 = threadIdx.x & 63;
    const size_t row = (size_t)blockIdx.x * 4 + r;
    const __bf16* q = qb + row * HEADSZ;
    float s = 0.f;
#pragma unroll
    for (int h = 0; h < HEADSZ; ++h)
        s += (float)q[h] * Ms[h * HEADSZ + h2];
    out[row * HEADSZ + h2] = s * 0.03608439182435161f;   // 768^-0.5
}

// ---------------------------------------------------------------------------
// Host launcher. Workspace layout (bytes):
//   [0, 2M)        qb   bf16 [16384,64]
//   [2M, 4M)       kb   bf16
//   [4M, 6M)       vb   bf16
//   [6M, 8M)       Pbuf f32 [4][32][64][64]
//   [8M, 8M+64K)   Mbuf f32 [4][64][64]
//   [.., +288K)    Wt   bf16 [192][768]  (prepacked transposed weights)
// ---------------------------------------------------------------------------
extern "C" void kernel_launch(void* const* d_in, const int* in_sizes, int n_in,
                              void* d_out, int out_size, void* d_ws, size_t ws_size,
                              hipStream_t stream)
{
    const float* x  = (const float*)d_in[0];
    const float* Wq = (const float*)d_in[1];
    const float* bq = (const float*)d_in[2];
    const float* Wk = (const float*)d_in[3];
    const float* bk = (const float*)d_in[4];
    const float* Wv = (const float*)d_in[5];
    const float* bv = (const float*)d_in[6];
    float* out = (float*)d_out;

    char* ws = (char*)d_ws;
    const size_t qkv_bytes = (size_t)NROWS * HEADSZ * sizeof(__bf16);   // 2 MB each
    __bf16* qb = (__bf16*)(ws);
    __bf16* kb = (__bf16*)(ws + qkv_bytes);
    __bf16* vb = (__bf16*)(ws + 2 * qkv_bytes);
    float*  Pbuf = (float*)(ws + 3 * qkv_bytes);
    float*  Mbuf = (float*)(ws + 3 * qkv_bytes +
                            (size_t)BATCH * NCHUNKS * HEADSZ * HEADSZ * sizeof(float));
    __bf16* Wt = (__bf16*)(ws + 3 * qkv_bytes +
                           (size_t)BATCH * NCHUNKS * HEADSZ * HEADSZ * sizeof(float) +
                           (size_t)BATCH * HEADSZ * HEADSZ * sizeof(float));

    prepack_w_kernel<<<dim3((NCOL * N_EMBD + 255) / 256), dim3(256), 0, stream>>>(
        Wq, Wk, Wv, Wt);

    proj_qkv_kernel<<<dim3(NROWS / MBLK), dim3(PROJ_THREADS), 0, stream>>>(
        x, Wt, bq, bk, bv, qb, kb, vb);

    ktv_partial_kernel<<<dim3(BATCH * NCHUNKS), dim3(256), 0, stream>>>(kb, vb, Pbuf);

    ktv_reduce_kernel<<<dim3((BATCH * HEADSZ * HEADSZ) / 256), dim3(256), 0, stream>>>(
        Pbuf, Mbuf);

    out_proj_kernel<<<dim3(NROWS / 4), dim3(256), 0, stream>>>(qb, Mbuf, out);
}
